// ControlledNODE_16673063043138
// MI455X (gfx1250) — matrix operand order repaired and checked by reading, not compiled, not run on hardware
//
#include <hip/hip_runtime.h>

// ------------------------------------------------------------------
// ControlledNODE scan on MI455X (gfx1250), single persistent wave32.
//
// Pure latency-chain workload (100000 sequential Euler steps, ~6.6K MACs
// each): one wave owns the problem; all weights resident in VGPRs as f16
// WMMA fragments. Layers 1/2 use vector-in-A (activation replicated across
// the 16 A-rows, W^T in B) -> output[n] lands in lane n. Layer 3 uses
// matrix-in-A (W3 resident in A, z2 broadcast in B) -> c[r] = dh[r+8*hi],
// exactly the layout of the persistent f32 state hreg[], so the state
// never touches LDS. Bias handling costs (almost) nothing:
//   - L1 + readout: x is K-padded 23->32, so x[23] is pinned to 1.0 and the
//     biases live in the weight fragments' k=23 row (zero extra VGPRs).
//   - L3: 8-reg bias-seeded C accumulator (feeds the state FMA directly).
//   - L2: plain VALU add (no spare K lane).
// Activations shuttle through 128B of LDS as f16 with table-order-K (L2 A) / contiguous-K (L3 B)
// fragments (2x ds_load_b128 per fragment). tanh is the gfx1250 hardware
// TRANS32 v_tanh_f32. u_t scalar loads are software-pipelined one step
// ahead. 15 WMMAs per step; scan loop un-unrolled to stay under 256 VGPRs.
// ------------------------------------------------------------------

#define T_STEPS 100000
#define HDIM 16
#define UDIM 7
#define XDIM 23
#define HID  64

typedef _Float16 v16h __attribute__((ext_vector_type(16)));
typedef _Float16 h8   __attribute__((ext_vector_type(8)));
typedef float    v8f  __attribute__((ext_vector_type(8)));

struct H16 { h8 lo; h8 hi; };
static __device__ __forceinline__ v16h cat16(h8 lo, h8 hi) {
  H16 s{lo, hi};
  return __builtin_bit_cast(v16h, s);
}

static __device__ __forceinline__ v8f wmma16(v16h a, v16h b, v8f c) {
  // D = A(16x32 f16) * B(32x16 f16) + C(16x16 f32)
  return __builtin_amdgcn_wmma_f32_16x16x32_f16(
      /*neg_a=*/false, a, /*neg_b=*/false, b,
      /*c_mod=*/(short)0, c, /*reuse_a=*/false, /*reuse_b=*/false);
}

static __device__ __forceinline__ float htanh(float x) {
  // gfx1250 hardware transcendental: one TRANS32 op, no branches.
  float r;
  asm("v_tanh_f32 %0, %1" : "=v"(r) : "v"(x));
  return r;
}

static __device__ __forceinline__ void ds_fence() {
  // single-wave LDS store->load ordering (hw is in-order per wave); this is
  // a compiler barrier + explicit counter wait before dependent reloads.
  asm volatile("s_wait_dscnt 0" ::: "memory");
}

__launch_bounds__(32)
__global__ void node_scan_kernel(
    const float* __restrict__ U,
    const float* __restrict__ W1, const float* __restrict__ b1,
    const float* __restrict__ W2, const float* __restrict__ b2,
    const float* __restrict__ W3, const float* __restrict__ b3,
    const float* __restrict__ wd, const float* __restrict__ bdp,
    const float* __restrict__ wt, const float* __restrict__ btp,
    const float* __restrict__ wc, const float* __restrict__ bcp,
    const float* __restrict__ h0,
    float* __restrict__ out)
{
  const int lane = (int)(threadIdx.x & 31);
  const int n    = lane & 15;   // output column (layers 1/2) / W3 row (layer 3)
  const int hi   = lane >> 4;   // which 16-lane half

  __shared__ __align__(16) _Float16 zlds[HID];  // inter-layer activations (f16)
  const char* zb = (const char*)zlds;

  // ---------------- resident fragments (f16) ----------------
  // B-frag (32x16): v16h elem e <-> B[k = e + 8*hi + 8*(e>=8), n = lane&15]
  // A-frag (16x32): v16h elem e <-> A[m = lane&15, k = e + 16*hi]
  v16h Br;        // readout rows 0..2 = wd,wt,wc; k=23 row carries bd/bt/bc
  v16h B1f[4];    // W1^T (64x23 -> K padded to 32); k=23 row carries b1
  v16h B2f[4][2]; // W2^T (64x64), [n-block b][k-chunk c]
  v16h A3w[2];    // W3 (16x64) as A-fragments, [k-chunk c]

  #pragma unroll
  for (int e = 0; e < 16; ++e) {
    int k = e + 8 * hi + ((e >= 8) ? 8 : 0);
    float w = 0.f;
    if (k < 16) {          // readout weight rows
      if      (n == 0) w = wd[k];
      else if (n == 1) w = wt[k];
      else if (n == 2) w = wc[k];
    } else if (k == 23) {  // bias via the pinned x[23]=1 lane
      if      (n == 0) w = bdp[0];
      else if (n == 1) w = btp[0];
      else if (n == 2) w = bcp[0];
    }
    Br[e] = (_Float16)w;
  }
  #pragma unroll
  for (int b = 0; b < 4; ++b) {
    #pragma unroll
    for (int e = 0; e < 16; ++e) {
      int k = e + 8 * hi + ((e >= 8) ? 8 : 0);
      float w;
      if      (k < XDIM)  w = W1[(16 * b + n) * XDIM + k];
      else if (k == XDIM) w = 0.f;  // reached for k=23; overwritten by the bias below
      else                w = 0.f;
      if (k == 23) w = b1[16 * b + n];      // bias via pinned x[23]=1
      B1f[b][e] = (_Float16)((k == 23) ? w
                   : (k < XDIM) ? W1[(16 * b + n) * XDIM + k] : 0.f);
    }
  }
  #pragma unroll
  for (int b = 0; b < 4; ++b) {
    #pragma unroll
    for (int c = 0; c < 2; ++c) {
      #pragma unroll
      for (int e = 0; e < 16; ++e) {
        int k = 32 * c + e + 8 * hi + ((e >= 8) ? 8 : 0);
        B2f[b][c][e] = (_Float16)W2[(16 * b + n) * HID + k];
      }
    }
  }
  #pragma unroll
  for (int c = 0; c < 2; ++c) {
    #pragma unroll
    for (int e = 0; e < 16; ++e) {
      int k = e + 16 * hi;   // A-frag K mapping (contiguous, matches z2 B loads)
      A3w[c][e] = (_Float16)W3[n * HID + 32 * c + k];
    }
  }

  // layer-2 bias: plain per-lane scalars (VALU add after WMMA)
  float b2l[4];
  #pragma unroll
  for (int b = 0; b < 4; ++b) b2l[b] = b2[16 * b + n];

  // layer-3 bias: seeded C accumulator, c[r] = b3[r+8hi] (state layout)
  v8f cb3;
  #pragma unroll
  for (int r = 0; r < 8; ++r) cb3[r] = b3[r + 8 * hi];

  const float dt = 5.0f / 60.0f;

  // ------------- persistent state: hreg[r] = h[r + 8*hi] (f32) -------------
  float hreg[8];
  #pragma unroll
  for (int r = 0; r < 8; ++r) hreg[r] = h0[r + 8 * hi];

  // ------------- software-pipelined u_t (wave-uniform scalar loads) --------
  // upk = f16 pack of x[16..23]; x[23] pinned to 1.0 (bias lane), hi lanes 0.
  float uu[7];
  #pragma unroll
  for (int i = 0; i < UDIM; ++i) uu[i] = U[i];
  h8 upk;
  #pragma unroll
  for (int e = 0; e < 8; ++e)
    upk[e] = hi ? (_Float16)0.f
                : (e < UDIM ? (_Float16)uu[e] : (_Float16)1.0f);

  const v8f zero8 = {};

  #pragma clang loop unroll(disable)
  for (int t = 0; t < T_STEPS; ++t) {
    // kick off next step's u loads right away (hide SMEM latency)
    const int tn = (t + 1 < T_STEPS) ? (t + 1) : t;
    float un[7];
    #pragma unroll
    for (int i = 0; i < UDIM; ++i) un[i] = U[7 * tn + i];
    __builtin_prefetch(U + 7 * t + 448, 0, 1);  // global_prefetch_b8, ~64 steps ahead

    // ---- A1 = x = [h ; u ; 1] replicated across rows ----
    h8 alo;
    #pragma unroll
    for (int e = 0; e < 8; ++e) alo[e] = (_Float16)hreg[e];  // h[e+8hi]
    v16h a1 = cat16(alo, upk);

    // ---- readouts (before Euler step): one WMMA, lanes 0/1/2 -> d/tx/ll ----
    {
      v8f cr = wmma16(a1, Br, zero8);        // bias folded into Br (k=23)
      if (lane < 3) out[lane * T_STEPS + t] = cr[0];
    }

    // ---- layer 1: z1 = tanh(W1 x + b1), lane n holds z1[16b+n] ----
    float z1[4];
    #pragma unroll
    for (int b = 0; b < 4; ++b) {
      v8f c = wmma16(a1, B1f[b], zero8);     // bias folded into B1f (k=23)
      z1[b] = htanh(c[0]);
    }
    #pragma unroll
    for (int b = 0; b < 4; ++b) zlds[16 * b + n] = (_Float16)z1[b]; // dup halves: same addr, same data
    ds_fence();

    // ---- A2 fragments: table-order K, 4x ds_load_b128 ----
    h8 p0 = *(const h8*)(zb + 16 * hi);        // z1[ 8hi ..  8hi+7]
    h8 p1 = *(const h8*)(zb + 32 + 16 * hi);   // z1[16+8hi .. 23+8hi]
    h8 p2 = *(const h8*)(zb + 64 + 16 * hi);   // z1[32+8hi ..]
    h8 p3 = *(const h8*)(zb + 96 + 16 * hi);   // z1[48+8hi ..]
    v16h a2lo = cat16(p0, p1);
    v16h a2hi = cat16(p2, p3);

    // ---- layer 2: z2 = tanh(W2 z1 + b2) ----
    float z2[4];
    #pragma unroll
    for (int b = 0; b < 4; ++b) {
      v8f c = wmma16(a2lo, B2f[b][0], zero8);
      c     = wmma16(a2hi, B2f[b][1], c);
      z2[b] = htanh(c[0] + b2l[b]);
    }
    #pragma unroll
    for (int b = 0; b < 4; ++b) zlds[16 * b + n] = (_Float16)z2[b];
    ds_fence();

    // ---- B fragments of z2 (broadcast columns): 4x ds_load_b128 ----
    h8 q0 = *(const h8*)(zb + 32 * hi);            // z2[16hi .. 16hi+7]
    h8 q1 = *(const h8*)(zb + 32 * hi + 16);       // z2[16hi+8 .. 16hi+15]
    h8 q2 = *(const h8*)(zb + 64 + 32 * hi);       // z2[32+16hi ..]
    h8 q3 = *(const h8*)(zb + 64 + 32 * hi + 16);
    v16h b3z0 = cat16(q0, q1);
    v16h b3z1 = cat16(q2, q3);

    // ---- layer 3 (matrix-in-A, bias pre-seeded) + Euler update ----
    {
      v8f c = wmma16(A3w[0], b3z0, cb3);
      c     = wmma16(A3w[1], b3z1, c);       // c[r] = dh[r + 8hi] + b3[r+8hi]
      #pragma unroll
      for (int r = 0; r < 8; ++r) hreg[r] += dt * c[r];   // single FMA each
    }

    // finish next step's u pack (off the critical path)
    #pragma unroll
    for (int e = 0; e < 8; ++e)
      upk[e] = hi ? (_Float16)0.f
                  : (e < UDIM ? (_Float16)un[e] : (_Float16)1.0f);
  }

  // final state hT -> out[3T .. 3T+15] (lanes 0 and 16 write their halves)
  if (n == 0) {
    #pragma unroll
    for (int r = 0; r < 8; ++r) out[3 * T_STEPS + 8 * hi + r] = hreg[r];
  }
}

extern "C" void kernel_launch(void* const* d_in, const int* in_sizes, int n_in,
                              void* d_out, int out_size, void* d_ws, size_t ws_size,
                              hipStream_t stream) {
  (void)in_sizes; (void)n_in; (void)out_size; (void)d_ws; (void)ws_size;
  const float* U  = (const float*)d_in[0];
  const float* W1 = (const float*)d_in[1];
  const float* b1 = (const float*)d_in[2];
  const float* W2 = (const float*)d_in[3];
  const float* b2 = (const float*)d_in[4];
  const float* W3 = (const float*)d_in[5];
  const float* b3 = (const float*)d_in[6];
  const float* wd = (const float*)d_in[7];
  const float* bd = (const float*)d_in[8];
  const float* wt = (const float*)d_in[9];
  const float* bt = (const float*)d_in[10];
  const float* wc = (const float*)d_in[11];
  const float* bc = (const float*)d_in[12];
  const float* h0 = (const float*)d_in[13];
  float* out = (float*)d_out;

  // Sequential scan: one persistent wave32 owns the whole problem.
  node_scan_kernel<<<dim3(1), dim3(32), 0, stream>>>(
      U, W1, b1, W2, b2, W3, b3, wd, bd, wt, bt, wc, bc, h0, out);
}
